// ResetGRU_90297392431152
// MI455X (gfx1250) — compile-verified, matrix operand order checked by reading, not run on hardware
//
#include <hip/hip_runtime.h>

#define LN_EPS 1e-5f

typedef __attribute__((ext_vector_type(16))) __bf16 bf16x16;
typedef __attribute__((ext_vector_type(8)))  float  v8f;
typedef __attribute__((ext_vector_type(8)))  int    v8i;
typedef __attribute__((ext_vector_type(4)))  int    i32x4;
typedef __attribute__((ext_vector_type(8)))  int    i32x8;
typedef __attribute__((ext_vector_type(4)))  unsigned int u32x4;

#if defined(__has_builtin)
#if __has_builtin(__builtin_amdgcn_tensor_load_to_lds) && __has_builtin(__builtin_amdgcn_s_wait_tensorcnt)
#define HAVE_TDM 1
#endif
#endif
#if defined(HAVE_TDM) && __has_include(<hip/amd_detail/amd_gfx1250_TDM.h>)
#define TDM_6ARG 1
#endif

// ---------- helpers ----------
__device__ __forceinline__ unsigned short f2bf(float f) {
    unsigned int u = __float_as_uint(f);
    u += 0x7FFFu + ((u >> 16) & 1u);          // round-to-nearest-even
    return (unsigned short)(u >> 16);
}
__device__ __forceinline__ float bf2f(unsigned short s) {
    return __uint_as_float(((unsigned int)s) << 16);
}
__device__ __forceinline__ bf16x16 load_frag(const unsigned short* p) {
    v8i t = *reinterpret_cast<const v8i*>(p);  // 32B contiguous global load
    return __builtin_bit_cast(bf16x16, t);
}
__device__ __forceinline__ bf16x16 load_frag_lds(const unsigned short* p) {
    i32x4 lo = *reinterpret_cast<const i32x4*>(p);       // 16B-aligned LDS reads
    i32x4 hi = *reinterpret_cast<const i32x4*>(p + 8);
    v8i t;
    t[0] = lo[0]; t[1] = lo[1]; t[2] = lo[2]; t[3] = lo[3];
    t[4] = hi[0]; t[5] = hi[1]; t[6] = hi[2]; t[7] = hi[3];
    return __builtin_bit_cast(bf16x16, t);
}
__device__ __forceinline__ float block_sum256(float v, float* sbuf) {
    int tid = threadIdx.x;
    sbuf[tid] = v;
    __syncthreads();
    for (int s = 128; s > 0; s >>= 1) {
        if (tid < s) sbuf[tid] += sbuf[tid + s];
        __syncthreads();
    }
    float r = sbuf[0];
    __syncthreads();
    return r;
}

// ---------- stage one 64-row x 256-elem bf16 weight tile into LDS (pitch 264) ----------
// TDM path: one tensor_load_to_lds descriptor, HW pad 4 dwords every 128 dwords -> pitch 264.
// Fallback: cooperative vector copy with the same padded layout.
__device__ __forceinline__ void stage_b_tile(const unsigned short* __restrict__ gsrc,
                                             unsigned short* lds, int row_stride_elems) {
#if defined(HAVE_TDM)
    if (threadIdx.x < 32) {   // one wave issues the DMA; EXEC is ignored by TDM
        unsigned long long ga = (unsigned long long)(const void*)gsrc;
        u32x4 g0;
        g0[0] = 1u;                                   // count=1 (valid user descriptor)
        g0[1] = (unsigned)(size_t)(void*)lds;         // lds_addr (flat low 32 = LDS offset)
        g0[2] = (unsigned)ga;                         // global_addr[31:0]
        g0[3] = (unsigned)((ga >> 32) & 0x1FFFFFFull) // global_addr[56:32]
                | (2u << 30);                         // type=2 ("image")
        i32x8 g1;
        g1[0] = (int)((1u << 16)      // data_size = 1 -> 2 bytes/elem
                    | (1u << 20)      // pad_enable
                    | (6u << 22)      // pad_interval: every 128 dwords
                    | (3u << 25));    // pad_amount: 4 dwords -> pitch 264 elems
        g1[1] = (int)(256u << 16);    // tensor_dim0 = 256 (bits 79:48, low16 here)
        g1[2] = (int)(64u << 16);     // tensor_dim0 hi=0 | tensor_dim1 = 64 (low16)
        g1[3] = (int)(256u << 16);    // tensor_dim1 hi=0 | tile_dim0 = 256
        g1[4] = 64;                   // tile_dim1 = 64, tile_dim2 = 0
        g1[5] = row_stride_elems;     // tensor_dim0_stride low 32
        g1[6] = 0;
        g1[7] = 0;
        i32x4 z4 = {0, 0, 0, 0};
#if defined(TDM_6ARG)
        i32x8 z8 = {0, 0, 0, 0, 0, 0, 0, 0};
        __builtin_amdgcn_tensor_load_to_lds(g0, g1, z4, z4, z8, 0);
#else
        __builtin_amdgcn_tensor_load_to_lds(g0, g1, z4, z4, 0);
#endif
        __builtin_amdgcn_s_wait_tensorcnt((short)0);
    }
#else
    for (int c = threadIdx.x; c < 64 * 32; c += (int)blockDim.x) {
        int row = c >> 5, cw = c & 31;   // 32 x 16B chunks per 256-elem row
        *reinterpret_cast<i32x4*>(lds + row * 264 + cw * 8) =
            *reinterpret_cast<const i32x4*>(gsrc + (size_t)row * row_stride_elems + cw * 8);
    }
#endif
}

// ---------- f32 -> bf16 conversion ----------
__global__ void cvt_bf16_kernel(const float* __restrict__ src,
                                unsigned short* __restrict__ dst, int n) {
    int i = blockIdx.x * blockDim.x + threadIdx.x;
    int stride = gridDim.x * blockDim.x;
    for (; i < n; i += stride) dst[i] = f2bf(src[i]);
}

// ---------- input GEMM: OUT[16384,3072] = Xbf[16384,512] * Wbf[3072,512]^T + bias ----------
// block = 128 threads (4 waves), tile 64(M) x 64(N); grid = (48, 256); B tile staged via TDM->LDS
__global__ void gemm_ih_kernel(const unsigned short* __restrict__ X,
                               const unsigned short* __restrict__ W,
                               const float* __restrict__ bias,
                               unsigned short* __restrict__ OUT) {
    const int K = 512, N = 3072, KC = 256, PITCH = 264;
    __shared__ unsigned short lds_b[64 * PITCH];
    int lane = threadIdx.x & 31;
    int wave = threadIdx.x >> 5;
    int m0 = blockIdx.y * 64 + wave * 16;
    int n0 = blockIdx.x * 64;
    int lm = lane & 15;
    int lk = (lane >> 4) * 16;
    const unsigned short* ap = X + (size_t)(m0 + lm) * K + lk;

    v8f acc[4] = {};
    for (int kc = 0; kc < K; kc += KC) {
        __syncthreads();
        stage_b_tile(W + (size_t)n0 * K + kc, lds_b, K);
        __syncthreads();
        const unsigned short* lb = lds_b + lk;
        for (int kk = 0; kk < KC; kk += 32) {
            __builtin_prefetch(ap + kc + kk + 64, 0, 1);
            bf16x16 a = load_frag(ap + kc + kk);
#pragma unroll
            for (int nt = 0; nt < 4; ++nt) {
                bf16x16 b = load_frag_lds(lb + (nt * 16 + lm) * PITCH + kk);
                acc[nt] = __builtin_amdgcn_wmma_f32_16x16x32_bf16(
                    false, a, false, b, (short)0, acc[nt], false, false);
            }
        }
    }
    int rbase = m0 + (lane >> 4) * 8;
#pragma unroll
    for (int nt = 0; nt < 4; ++nt) {
        int col = n0 + nt * 16 + lm;
        float bv = bias[col];
#pragma unroll
        for (int j = 0; j < 8; ++j)
            OUT[(size_t)(rbase + j) * N + col] = f2bf(acc[nt][j] + bv);
    }
}

// ---------- recurrent GEMM: OUT[64,3072] = Hbf[64,1024] * Wbf[3072,1024]^T + bias ----------
// grid = (48,1), block = 128 (4 waves cover all 64 batch rows); B tile staged via TDM->LDS
__global__ void gemm_hh_kernel(const unsigned short* __restrict__ Hb,
                               const unsigned short* __restrict__ W,
                               const float* __restrict__ bias,
                               float* __restrict__ OUT) {
    const int K = 1024, N = 3072, KC = 256, PITCH = 264;
    __shared__ unsigned short lds_b[64 * PITCH];
    int lane = threadIdx.x & 31;
    int wave = threadIdx.x >> 5;
    int m0 = wave * 16;
    int n0 = blockIdx.x * 64;
    int lm = lane & 15;
    int lk = (lane >> 4) * 16;
    const unsigned short* ap = Hb + (size_t)(m0 + lm) * K + lk;

    v8f acc[4] = {};
    for (int kc = 0; kc < K; kc += KC) {
        __syncthreads();
        stage_b_tile(W + (size_t)n0 * K + kc, lds_b, K);
        __syncthreads();
        const unsigned short* lb = lds_b + lk;
        for (int kk = 0; kk < KC; kk += 32) {
            __builtin_prefetch(ap + kc + kk + 64, 0, 1);
            bf16x16 a = load_frag(ap + kc + kk);
#pragma unroll
            for (int nt = 0; nt < 4; ++nt) {
                bf16x16 b = load_frag_lds(lb + (nt * 16 + lm) * PITCH + kk);
                acc[nt] = __builtin_amdgcn_wmma_f32_16x16x32_bf16(
                    false, a, false, b, (short)0, acc[nt], false, false);
            }
        }
    }
    int rbase = m0 + (lane >> 4) * 8;
#pragma unroll
    for (int nt = 0; nt < 4; ++nt) {
        int col = n0 + nt * 16 + lm;
        float bv = bias[col];
#pragma unroll
        for (int j = 0; j < 8; ++j)
            OUT[(size_t)(rbase + j) * N + col] = acc[nt][j] + bv;
    }
}

// ---------- in-place LN of ih rows: outer LN over 3072, inner LN over n-slice (cols 2048..3071) ----------
// grid = 16384 (one row per block), block = 256
__global__ void ln_ih_kernel(unsigned short* __restrict__ IH,
                             const float* __restrict__ g1, const float* __restrict__ b1,
                             const float* __restrict__ g2, const float* __restrict__ b2) {
    __shared__ float sbuf[256];
    const int N = 3072;
    int tid = threadIdx.x;
    unsigned short* prow = IH + (size_t)blockIdx.x * N;

    float v[12];
#pragma unroll
    for (int k = 0; k < 12; ++k) v[k] = bf2f(prow[tid + 256 * k]);

    float s = 0.f, ss = 0.f;
#pragma unroll
    for (int k = 0; k < 12; ++k) { s += v[k]; ss += v[k] * v[k]; }
    s = block_sum256(s, sbuf);
    ss = block_sum256(ss, sbuf);
    float mean = s * (1.f / N);
    float inv = rsqrtf(ss * (1.f / N) - mean * mean + LN_EPS);
#pragma unroll
    for (int k = 0; k < 12; ++k) {
        int col = tid + 256 * k;
        v[k] = (v[k] - mean) * inv * g1[col] + b1[col];
    }
    float s2 = 0.f, ss2 = 0.f;
#pragma unroll
    for (int k = 8; k < 12; ++k) { s2 += v[k]; ss2 += v[k] * v[k]; }
    s2 = block_sum256(s2, sbuf);
    ss2 = block_sum256(ss2, sbuf);
    float m2 = s2 * (1.f / 1024.f);
    float inv2 = rsqrtf(ss2 * (1.f / 1024.f) - m2 * m2 + LN_EPS);

#pragma unroll
    for (int k = 0; k < 8; ++k) prow[tid + 256 * k] = f2bf(v[k]);
#pragma unroll
    for (int k = 8; k < 12; ++k) {
        int q = tid + 256 * (k - 8);
        prow[tid + 256 * k] = f2bf((v[k] - m2) * inv2 * g2[q] + b2[q]);
    }
}

// ---------- initial reset blend for t=0 ----------
__global__ void blend0_kernel(const float* __restrict__ state,
                              const float* __restrict__ reset,
                              const int* __restrict__ idx,
                              const float* __restrict__ init,
                              float* __restrict__ hf, unsigned short* __restrict__ hb) {
    int i = blockIdx.x * 256 + threadIdx.x;   // 0..65535
    int b = i >> 10, q = i & 1023;
    float r = reset[b];                        // reset[0, b]
    int s = idx[b];                            // idx[0, b]
    float v = state[i] * (1.f - r) + init[s * 1024 + q] * r;
    hf[i] = v;
    hb[i] = f2bf(v);
}

// ---------- per-step fused: LN(hh) + inner LN + gates + output + next-step reset blend ----------
// grid = 64 (one batch row per block), block = 256
__global__ void gru_step_kernel(const float* __restrict__ HH,            // [64,3072]
                                const unsigned short* __restrict__ IH,   // [16384,3072]
                                const float* __restrict__ ghh, const float* __restrict__ bhh,
                                const float* __restrict__ ghn, const float* __restrict__ bhn,
                                const float* __restrict__ reset, const int* __restrict__ idx,
                                const float* __restrict__ init,
                                float* __restrict__ hf, unsigned short* __restrict__ hb,
                                float* __restrict__ out, int t) {
    __shared__ float sbuf[256];
    const int N = 3072, B = 64, T = 256, H = 1024;
    int b = blockIdx.x, tid = threadIdx.x;
    const float* hrow = HH + (size_t)b * N;

    float v[12];
#pragma unroll
    for (int k = 0; k < 12; ++k) v[k] = hrow[tid + 256 * k];

    float s = 0.f, ss = 0.f;
#pragma unroll
    for (int k = 0; k < 12; ++k) { s += v[k]; ss += v[k] * v[k]; }
    s = block_sum256(s, sbuf);
    ss = block_sum256(ss, sbuf);
    float mean = s * (1.f / N);
    float inv = rsqrtf(ss * (1.f / N) - mean * mean + LN_EPS);
#pragma unroll
    for (int k = 0; k < 12; ++k) {
        int col = tid + 256 * k;
        v[k] = (v[k] - mean) * inv * ghh[col] + bhh[col];
    }
    float s2 = 0.f, ss2 = 0.f;
#pragma unroll
    for (int k = 8; k < 12; ++k) { s2 += v[k]; ss2 += v[k] * v[k]; }
    s2 = block_sum256(s2, sbuf);
    ss2 = block_sum256(ss2, sbuf);
    float m2 = s2 * (1.f / 1024.f);
    float inv2 = rsqrtf(ss2 * (1.f / 1024.f) - m2 * m2 + LN_EPS);
#pragma unroll
    for (int k = 8; k < 12; ++k) {
        int q = tid + 256 * (k - 8);
        v[k] = (v[k] - m2) * inv2 * ghn[q] + bhn[q];
    }

    const unsigned short* irow = IH + ((size_t)t * B + b) * N;
    float rn = 0.f;
    int sn = 0;
    if (t + 1 < T) { rn = reset[(t + 1) * B + b]; sn = idx[(t + 1) * B + b]; }

#pragma unroll
    for (int kq = 0; kq < 4; ++kq) {
        int q = tid + 256 * kq;
        float ir = bf2f(irow[q]);
        float iz = bf2f(irow[1024 + q]);
        float inn = bf2f(irow[2048 + q]);
        float r = 1.f / (1.f + __expf(-(ir + v[kq])));
        float z = 1.f / (1.f + __expf(-(iz + v[kq + 4])));
        float n = tanhf(inn + r * v[kq + 8]);
        float hprev = hf[b * H + q];
        float hnew = (1.f - z) * n + z * hprev;
        out[((size_t)t * B + b) * H + q] = hnew;
        if (t == T - 1) out[(size_t)T * B * H + b * H + q] = hnew;
        float hbl = hnew;
        if (t + 1 < T) hbl = hnew * (1.f - rn) + init[sn * H + q] * rn;
        hf[b * H + q] = hbl;
        hb[b * H + q] = f2bf(hbl);
    }
}

// ---------- launch ----------
extern "C" void kernel_launch(void* const* d_in, const int* in_sizes, int n_in,
                              void* d_out, int out_size, void* d_ws, size_t ws_size,
                              hipStream_t stream) {
    (void)in_sizes; (void)n_in; (void)out_size; (void)ws_size;
    const float* x      = (const float*)d_in[0];
    const float* state  = (const float*)d_in[1];
    const float* reset  = (const float*)d_in[2];
    const int*   idx    = (const int*)d_in[3];
    const float* wih    = (const float*)d_in[4];
    const float* whh    = (const float*)d_in[5];
    const float* bih    = (const float*)d_in[6];
    const float* bhh    = (const float*)d_in[7];
    const float* gih    = (const float*)d_in[8];
    const float* bihl   = (const float*)d_in[9];
    const float* ghh    = (const float*)d_in[10];
    const float* bhhl   = (const float*)d_in[11];
    const float* gin    = (const float*)d_in[12];
    const float* bin    = (const float*)d_in[13];
    const float* ghn    = (const float*)d_in[14];
    const float* bhn    = (const float*)d_in[15];
    const float* init   = (const float*)d_in[16];
    float* out = (float*)d_out;

    char* ws = (char*)d_ws;
    size_t off = 0;
    auto alloc = [&](size_t bytes) -> void* {
        void* p = ws + off;
        off = (off + bytes + 255) & ~(size_t)255;
        return p;
    };
    unsigned short* xb   = (unsigned short*)alloc(16384ull * 512 * 2);
    unsigned short* wihb = (unsigned short*)alloc(3072ull * 512 * 2);
    unsigned short* whhb = (unsigned short*)alloc(3072ull * 1024 * 2);
    unsigned short* ihb  = (unsigned short*)alloc(16384ull * 3072 * 2);
    float*          hf   = (float*)alloc(64ull * 1024 * 4);
    unsigned short* hb   = (unsigned short*)alloc(64ull * 1024 * 2);
    float*          hh   = (float*)alloc(64ull * 3072 * 4);

    cvt_bf16_kernel<<<4096, 256, 0, stream>>>(x,   xb,   16384 * 512);
    cvt_bf16_kernel<<<1024, 256, 0, stream>>>(wih, wihb, 3072 * 512);
    cvt_bf16_kernel<<<2048, 256, 0, stream>>>(whh, whhb, 3072 * 1024);

    gemm_ih_kernel<<<dim3(48, 256), 128, 0, stream>>>(xb, wihb, bih, ihb);
    ln_ih_kernel<<<16384, 256, 0, stream>>>(ihb, gih, bihl, gin, bin);
    blend0_kernel<<<256, 256, 0, stream>>>(state, reset, idx, init, hf, hb);

    for (int t = 0; t < 256; ++t) {
        gemm_hh_kernel<<<dim3(48, 1), 128, 0, stream>>>(hb, whhb, bhh, hh);
        gru_step_kernel<<<64, 256, 0, stream>>>(hh, ihb, ghh, bhhl, ghn, bhn,
                                                reset, idx, init, hf, hb, out, t);
    }
}
